// CIGINModel_57629871178232
// MI455X (gfx1250) — matrix operand order repaired and checked by reading, not compiled, not run on hardware
//
#include <hip/hip_runtime.h>

// CIGIN on gfx1250: per-edge NNConv matvec restructured into dense bf16 WMMA GEMMs
//   T = out @ W2aug  (W2aug[i, k*42+o] = e2w[k,i*42+o]; e2b folded in as k=10 channel)
//   msg[e,o] = sum_k h1[e,k] * T[src[e],k,o];  scatter-add via fp32 atomics
// WMMA operands are stored PRE-SWIZZLED in the per-lane register layout so each
// wave loads its A tile as 64 contiguous bytes/lane and each B k-step as one
// contiguous v16bf/lane (all b128, fully coalesced). All working buffers are
// L2-resident; HBM traffic ~= the 1.34GB zero interaction output (memset).

#define DD   42       // node hidden dim
#define EDIM 10       // edge feature dim
#define DP   48       // padded feature width (fp32 buffers), 3x16
#define KP   64       // padded K for bf16 A buffers, 2x32
#define KE   11       // edge-net channels incl. bias channel
#define TW   462      // KE * DD
#define TWP  464      // padded to 16
#define MT   4        // M-tiles per wave in the GEMM

static constexpr int NU = 16384, EU_ = 32768, NV = 20480, EV_ = 40960;
static constexpr int NMAX = 20480, EMAX = 40960;

typedef __attribute__((ext_vector_type(16))) __bf16 v16bf;
typedef __attribute__((ext_vector_type(8)))  float  v8f;

__device__ __forceinline__ float frelu(float x) { return x > 0.f ? x : 0.f; }

// ---- swizzled storage layouts (match WMMA VGPR layouts, ISA 7.12.2) --------
// A tile (16x64 bf16, one 16-row M-tile): lane = kh*16 + (m&15) holds 32
// contiguous elements; element pos = g*8+j  <->  K = g*16 + kh*8 + j.
__device__ __forceinline__ size_t a_swz_idx(int n, int o) {
  int t = n >> 4, mrow = n & 15;
  int g = o >> 4, w = o & 15;
  int kh = w >> 3, j = w & 7;
  return (size_t)t * 1024 + (size_t)(kh * 16 + mrow) * 32 + g * 8 + j;
}
// B k-step chunk (32x16 bf16): lane = ((K>>3)&1)*16 + (col&15) holds 16
// elements; element j = ((K>>4)&1)*8 + (K&7); chunk index = ntile*2 + (K>>5).
__device__ __forceinline__ size_t b_swz_idx(int K, int col) {
  int ntile = col >> 4, n = col & 15;
  int s = K >> 5, hi = (K >> 3) & 1, j = ((K >> 4) & 1) * 8 + (K & 7);
  return ((size_t)(ntile * 2 + s) * 32 + (hi * 16 + n)) * 16 + j;
}

// ---------------------------------------------------------------------------
// WMMA GEMM: C[M, ldc] (+bias,relu) = A[M, KP]bf16 @ B[KP, ncols]bf16
// One wave per (16-col tile x 4 M-tiles). K = 64 = two 16x16x32 bf16 WMMAs.
// A and B are pre-swizzled; all operand loads are contiguous b128.
// ---------------------------------------------------------------------------
template <bool BIAS_RELU, bool WRITE_BF>
__global__ __launch_bounds__(32) void wmma_gemm(
    const __bf16* __restrict__ A, const __bf16* __restrict__ B,
    float* __restrict__ C, int ldc, __bf16* __restrict__ Cbf,
    const float* __restrict__ bias) {
  const int lane = threadIdx.x;
  const int nt   = blockIdx.x;
  const int mt0  = blockIdx.y * MT;

  const v16bf* Bp = (const v16bf*)B;
  const v16bf b0 = Bp[(nt * 2 + 0) * 32 + lane];
  const v16bf b1 = Bp[(nt * 2 + 1) * 32 + lane];

  const int col = nt * 16 + (lane & 15);
  float bv = 0.f;
  if (BIAS_RELU) bv = bias[col];

#pragma unroll
  for (int mi = 0; mi < MT; ++mi) {
    const int mt = mt0 + mi;
    const v16bf* Ap = (const v16bf*)(A + (size_t)mt * 1024);
    if (mi + 1 < MT)
      __builtin_prefetch(A + (size_t)(mt + 1) * 1024 + lane * 32, 0, 3);
    const v16bf a0 = Ap[lane * 2];
    const v16bf a1 = Ap[lane * 2 + 1];

    v8f c = {0.f, 0.f, 0.f, 0.f, 0.f, 0.f, 0.f, 0.f};
    c = __builtin_amdgcn_wmma_f32_16x16x32_bf16(false, a0, false, b0, (short)0, c, false, false);
    c = __builtin_amdgcn_wmma_f32_16x16x32_bf16(false, a1, false, b1, (short)0, c, false, false);

    // C/D layout: VGPR r holds M = r + (lane>=16 ? 8 : 0), N = lane&15
    const int mb = mt * 16 + ((lane >> 4) ? 8 : 0);
#pragma unroll
    for (int r = 0; r < 8; ++r) {
      float v = c[r];
      if (BIAS_RELU) v = frelu(v + bv);
      C[(size_t)(mb + r) * ldc + col] = v;
      if (WRITE_BF) Cbf[a_swz_idx(mb + r, col)] = (__bf16)v;
    }
  }
}

// ---------------------------------------------------------------------------
// Weight / buffer preparation (write directly into swizzled layouts)
// ---------------------------------------------------------------------------
__global__ void k_pad_w48(const float* __restrict__ w, __bf16* __restrict__ dst) {
  int idx = blockIdx.x * blockDim.x + threadIdx.x;       // KP*DP
  if (idx >= KP * DP) return;
  int i = idx / DP, o = idx % DP;
  float v = (i < DD && o < DD) ? w[i * DD + o] : 0.f;
  dst[b_swz_idx(i, o)] = (__bf16)v;
}

__global__ void k_build_w2aug(const float* __restrict__ e2w, const float* __restrict__ e2b,
                              __bf16* __restrict__ dst) {
  int idx = blockIdx.x * blockDim.x + threadIdx.x;       // KP*TWP
  if (idx >= KP * TWP) return;
  int i = idx / TWP, col = idx % TWP;
  float v = 0.f;
  if (i < DD && col < TW) {
    int k = col / DD, o = col % DD;
    v = (k < EDIM) ? e2w[(size_t)k * (DD * DD) + i * DD + o] : e2b[i * DD + o];
  }
  dst[b_swz_idx(i, col)] = (__bf16)v;
}

__global__ void k_pad_bias(const float* __restrict__ b, float* __restrict__ dst) {
  int o = threadIdx.x;
  if (o < DP) dst[o] = (o < DD) ? b[o] : 0.f;
}

__global__ void k_edge_h1(const float* __restrict__ ef, const float* __restrict__ w1,
                          const float* __restrict__ b1, float* __restrict__ h1a, int E) {
  int e = blockIdx.x * blockDim.x + threadIdx.x;
  if (e >= E) return;
  float x[EDIM];
#pragma unroll
  for (int i = 0; i < EDIM; ++i) x[i] = ef[(size_t)e * EDIM + i];
#pragma unroll
  for (int j = 0; j < EDIM; ++j) {
    float acc = b1[j];
#pragma unroll
    for (int i = 0; i < EDIM; ++i) acc += x[i] * w1[i * EDIM + j];
    h1a[(size_t)e * 12 + j] = frelu(acc);
  }
  h1a[(size_t)e * 12 + 10] = 1.f;   // constant channel carrying e2b term
  h1a[(size_t)e * 12 + 11] = 0.f;
}

__global__ void k_pack_x(const float* __restrict__ x, __bf16* __restrict__ dst, int N) {
  int idx = blockIdx.x * blockDim.x + threadIdx.x;       // N*KP
  if (idx >= N * KP) return;
  int n = idx / KP, o = idx % KP;
  dst[a_swz_idx(n, o)] = (__bf16)((o < DD) ? x[(size_t)n * DD + o] : 0.f);
}

// msg[e,o] = sum_k h1[e,k] * T[src[e], k*42+o];  agg[dst[e], o] += msg  (one wave/edge)
__global__ __launch_bounds__(256) void k_edge_msg(
    const int* __restrict__ src, const int* __restrict__ dst,
    const float* __restrict__ h1a, const float* __restrict__ T,
    float* __restrict__ agg, int E) {
  int e = blockIdx.x * 8 + (threadIdx.x >> 5);
  if (e >= E) return;
  int lane = threadIdx.x & 31;
  int s = src[e], d = dst[e];
  float h[KE];
#pragma unroll
  for (int k = 0; k < KE; ++k) h[k] = h1a[(size_t)e * 12 + k];
  const float* Tr = T + (size_t)s * TWP;
  for (int o = lane; o < DD; o += 32) {
    float acc = 0.f;
#pragma unroll
    for (int k = 0; k < KE; ++k) acc += h[k] * Tr[k * DD + o];
    atomicAdd(&agg[(size_t)d * DP + o], acc);
  }
}

// m = relu(agg + out + cb) -> swizzled bf16 A-buffer (zero-padded to KP cols)
__global__ void k_node_m(const float* __restrict__ agg, const float* __restrict__ out,
                         const float* __restrict__ cb, __bf16* __restrict__ mbuf, int N) {
  int idx = blockIdx.x * blockDim.x + threadIdx.x;       // N*KP
  if (idx >= N * KP) return;
  int n = idx / KP, o = idx % KP;
  float v = 0.f;
  if (o < DD) v = frelu(agg[(size_t)n * DP + o] + out[(size_t)n * DP + o] + cb[o]);
  mbuf[a_swz_idx(n, o)] = (__bf16)v;
}

// column sums of (out + x_init); equal-sized graphs => global mean == mean of graph means
__global__ __launch_bounds__(64) void k_mean_acc(
    const float* __restrict__ out, const float* __restrict__ x,
    float* __restrict__ accum, int N) {
  int o = threadIdx.x;
  if (o >= DD) return;
  int r0 = blockIdx.x * 64;
  float acc = 0.f;
  for (int r = 0; r < 64; ++r) {
    int n = r0 + r;
    acc += out[(size_t)n * DP + o] + x[(size_t)n * DD + o];
  }
  atomicAdd(&accum[o], acc);
}

__global__ __launch_bounds__(256) void k_mlp(
    const float* __restrict__ accum,            // [84]: su sums | sv sums
    const float* __restrict__ fc1w, const float* __restrict__ fc1b,
    const float* __restrict__ fc2w, const float* __restrict__ fc2b,
    const float* __restrict__ fc3w, const float* __restrict__ fc3b,
    float* __restrict__ outp) {
  __shared__ float comb[4 * DD];
  __shared__ float h1s[256];
  __shared__ float h2s[128];
  int t = threadIdx.x;
  if (t < 4 * DD) {
    int half = t / (2 * DD);                    // 0: solute, 1: solvent
    int o = (t % (2 * DD)) % DD;
    comb[t] = half ? accum[DD + o] * (1.f / (float)NV)
                   : accum[o]      * (1.f / (float)NU);
  }
  __syncthreads();
  {
    float acc = fc1b[t];
    for (int i = 0; i < 4 * DD; ++i) acc += comb[i] * fc1w[i * 256 + t];
    h1s[t] = frelu(acc);
  }
  __syncthreads();
  if (t < 128) {
    float acc = fc2b[t];
    for (int i = 0; i < 256; ++i) acc += h1s[i] * fc2w[i * 128 + t];
    h2s[t] = frelu(acc);
  }
  __syncthreads();
  if (t == 0) {
    float acc = fc3b[0];
    for (int i = 0; i < 128; ++i) acc += h2s[i] * fc3w[i];
    outp[0] = acc;
  }
}

// ---------------------------------------------------------------------------
extern "C" void kernel_launch(void* const* d_in, const int* in_sizes, int n_in,
                              void* d_out, int out_size, void* d_ws, size_t ws_size,
                              hipStream_t stream) {
  (void)in_sizes; (void)n_in; (void)out_size; (void)ws_size;
  char* ws = (char*)d_ws;
  size_t off = 0;
  auto alloc = [&](size_t bytes) {
    off = (off + 255) & ~(size_t)255;
    size_t o = off; off += bytes; return o;
  };
  float*  out_f = (float*)(ws + alloc((size_t)NMAX * DP * 4));
  float*  agg_f = (float*)(ws + alloc((size_t)NMAX * DP * 4));
  __bf16* ax    = (__bf16*)(ws + alloc((size_t)NMAX * KP * 2));   // x / m (GEMM A, swizzled)
  __bf16* ao    = (__bf16*)(ws + alloc((size_t)NMAX * KP * 2));   // out (GEMM A, swizzled)
  float*  Tbuf  = (float*)(ws + alloc((size_t)NMAX * TWP * 4));
  float*  h1a   = (float*)(ws + alloc((size_t)EMAX * 12 * 4));
  __bf16* w2    = (__bf16*)(ws + alloc((size_t)KP * TWP * 2));    // swizzled B
  __bf16* lwb   = (__bf16*)(ws + alloc((size_t)KP * DP * 2));     // swizzled B
  __bf16* mwb   = (__bf16*)(ws + alloc((size_t)KP * DP * 2));     // swizzled B
  float*  lbp   = (float*)(ws + alloc(DP * 4));
  float*  mbp   = (float*)(ws + alloc(DP * 4));
  float*  cbp   = (float*)(ws + alloc(DP * 4));
  float*  accum = (float*)(ws + alloc(2 * DD * 4));

  const float* su_x  = (const float*)d_in[0];
  const float* su_e  = (const float*)d_in[1];
  const int*   su_s  = (const int*)d_in[2];
  const int*   su_d  = (const int*)d_in[3];
  const float* sv_x  = (const float*)d_in[4];
  const float* sv_e  = (const float*)d_in[5];
  const int*   sv_s  = (const int*)d_in[6];
  const int*   sv_d  = (const int*)d_in[7];
  const float* W[18];
  for (int i = 0; i < 18; ++i) W[i] = (const float*)d_in[8 + i];   // su:0..8 sv:9..17
  const float* fc1w = (const float*)d_in[26];
  const float* fc1b = (const float*)d_in[27];
  const float* fc2w = (const float*)d_in[28];
  const float* fc2b = (const float*)d_in[29];
  const float* fc3w = (const float*)d_in[30];
  const float* fc3b = (const float*)d_in[31];

  hipMemsetAsync(accum, 0, 2 * DD * 4, stream);
  // interaction output: all zeros, 1.34 GB -> dominant HBM traffic, pure memset
  hipMemsetAsync((float*)d_out + 1, 0, (size_t)NU * NV * 4, stream);

  auto run = [&](const float* x, const float* ef, const int* src, const int* dst,
                 const float* const* Wp, int N, int E, float* acc) {
    const float *lw = Wp[0], *lb = Wp[1], *mw = Wp[2], *mb = Wp[3];
    const float *e1w = Wp[4], *e1b = Wp[5], *e2w = Wp[6], *e2b = Wp[7], *cb = Wp[8];
    k_pad_w48<<<(KP * DP + 255) / 256, 256, 0, stream>>>(lw, lwb);
    k_pad_w48<<<(KP * DP + 255) / 256, 256, 0, stream>>>(mw, mwb);
    k_build_w2aug<<<(KP * TWP + 255) / 256, 256, 0, stream>>>(e2w, e2b, w2);
    k_pad_bias<<<1, DP, 0, stream>>>(lb, lbp);
    k_pad_bias<<<1, DP, 0, stream>>>(mb, mbp);
    k_pad_bias<<<1, DP, 0, stream>>>(cb, cbp);
    k_edge_h1<<<(E + 255) / 256, 256, 0, stream>>>(ef, e1w, e1b, h1a, E);
    int nkp_blocks = (int)(((size_t)N * KP + 255) / 256);
    k_pack_x<<<nkp_blocks, 256, 0, stream>>>(x, ax, N);
    // lin0: out = relu(x @ lw + lb)
    wmma_gemm<true, true><<<dim3(DP / 16, N / (16 * MT)), 32, 0, stream>>>(
        ax, lwb, out_f, DP, ao, lbp);
    for (int s6 = 0; s6 < 6; ++s6) {
      // T = out @ W2aug  (the heavy WMMA GEMM; T stays L2-resident)
      wmma_gemm<false, false><<<dim3(TWP / 16, N / (16 * MT)), 32, 0, stream>>>(
          ao, w2, Tbuf, TWP, nullptr, nullptr);
      hipMemsetAsync(agg_f, 0, (size_t)N * DP * 4, stream);
      k_edge_msg<<<(E + 7) / 8, 256, 0, stream>>>(src, dst, h1a, Tbuf, agg_f, E);
      k_node_m<<<nkp_blocks, 256, 0, stream>>>(agg_f, out_f, cbp, ax, N);
      // out = relu(m @ mw + mb)
      wmma_gemm<true, true><<<dim3(DP / 16, N / (16 * MT)), 32, 0, stream>>>(
          ax, mwb, out_f, DP, ao, mbp);
    }
    k_mean_acc<<<N / 64, 64, 0, stream>>>(out_f, x, acc, N);
  };

  run(su_x, su_e, su_s, su_d, W,     NU, EU_, accum);
  run(sv_x, sv_e, sv_s, sv_d, W + 9, NV, EV_, accum + DD);

  k_mlp<<<1, 256, 0, stream>>>(accum, fc1w, fc1b, fc2w, fc2b, fc3w, fc3b,
                               (float*)d_out);
}
